// GATEncoder_39548058862202
// MI455X (gfx1250) — matrix operand order repaired and checked by reading, not compiled, not run on hardware
//
#include <hip/hip_runtime.h>

typedef __attribute__((ext_vector_type(16))) _Float16 v16h;
typedef __attribute__((ext_vector_type(8)))  _Float16 v8h;
typedef __attribute__((ext_vector_type(8)))  float    v8f;

#define NEG_SLOPE 0.2f

#define V16_FROM_2xV8(lo, hi) \
    __builtin_shufflevector((lo), (hi), 0,1,2,3,4,5,6,7,8,9,10,11,12,13,14,15)

// ---------------------------------------------------------------------------
// Utility kernels
// ---------------------------------------------------------------------------
__global__ void k_f32_to_f16(const float* __restrict__ in, _Float16* __restrict__ out, int n) {
    int i = blockIdx.x * blockDim.x + threadIdx.x;
    if (i < n) out[i] = (_Float16)in[i];
}

__global__ void k_zero_f32(float* __restrict__ p, int n) {
    int i = blockIdx.x * blockDim.x + threadIdx.x;
    if (i < n) p[i] = 0.0f;
}

__device__ __forceinline__ void atomicMaxFloat(float* addr, float val) {
    if (val >= 0.0f) atomicMax((int*)addr, __float_as_int(val));
    else             atomicMin((unsigned int*)addr, __float_as_uint(val));
}

// ---------------------------------------------------------------------------
// Pack B[K,Nc] (f32 row-major) into WMMA-fragment order (f16):
//   Bp[((nt*Ks + ks)*32 + lane)*16 + j] = B[(ks*32 + (lane>=16?16:0) + j)*Nc + nt*16 + (lane&15)]
// so each lane's 16 B-halves for one 16x16 tile / 32-K step are contiguous (32B).
// ---------------------------------------------------------------------------
__global__ void k_pack_b_f16(const float* __restrict__ B, _Float16* __restrict__ Bp,
                             int K, int Nc) {
    int idx = blockIdx.x * blockDim.x + threadIdx.x;
    if (idx >= K * Nc) return;
    int Ks = K >> 5;
    int j    = idx & 15;
    int lane = (idx >> 4) & 31;
    int bi   = idx >> 9;           // (nt*Ks + ks)
    int ks   = bi % Ks;
    int nt   = bi / Ks;
    int k = ks * 32 + ((lane & 16) ? 16 : 0) + j;
    int n = nt * 16 + (lane & 15);
    Bp[idx] = (_Float16)B[(size_t)k * Nc + n];
}

// ---------------------------------------------------------------------------
// WMMA f16->f32 GEMM: C[M,Nc] = A[M,K] * B[K,Nc]; B pre-packed (k_pack_b_f16).
// One wave computes a 16x64 tile (4 accumulators, shared A fragment).
// Software-pipelined: step k+1's 10 b128 loads are issued before step k's
// 4 WMMAs consume the already-resident fragments.
// Requires M%16==0, Nc%64==0, K%32==0. EXEC all-ones (no divergence).
// ---------------------------------------------------------------------------
__global__ __launch_bounds__(32)
void k_wmma_gemm_f16(const _Float16* __restrict__ A, const _Float16* __restrict__ Bp,
                     float* __restrict__ C, int M, int Nc, int K) {
    const int mt   = blockIdx.x;
    const int nt4  = blockIdx.y;           // group of 4 col-tiles (64 cols)
    const int lane = threadIdx.x;          // 0..31
    const int r16  = lane & 15;
    const bool hi  = lane >= 16;

    const int m   = mt * 16 + r16;
    const int kbA = hi ? 16 : 0;
    const int Ks  = K >> 5;

    const _Float16* pa = A + (size_t)m * K + kbA;          // + ks*32 per step
    const _Float16* pb[4];
    #pragma unroll
    for (int q = 0; q < 4; ++q)                             // + ks*512 per step
        pb[q] = Bp + ((size_t)((nt4 * 4 + q) * Ks) * 32 + lane) * 16;

    v8f acc[4] = {};
    // ---- prologue: load fragments for ks = 0 ----
    v8h al = *(const v8h*)(pa);
    v8h ah = *(const v8h*)(pa + 8);
    v8h bl[4], bh[4];
    #pragma unroll
    for (int q = 0; q < 4; ++q) {
        bl[q] = *(const v8h*)(pb[q]);
        bh[q] = *(const v8h*)(pb[q] + 8);
    }

    // ---- steady state ----
    for (int ks = 0; ks + 1 < Ks; ++ks) {
        const int kn = ks + 1;
        v8h nal = *(const v8h*)(pa + (size_t)kn * 32);
        v8h nah = *(const v8h*)(pa + (size_t)kn * 32 + 8);
        v8h nbl[4], nbh[4];
        #pragma unroll
        for (int q = 0; q < 4; ++q) {
            nbl[q] = *(const v8h*)(pb[q] + (size_t)kn * 512);
            nbh[q] = *(const v8h*)(pb[q] + (size_t)kn * 512 + 8);
        }
        v16h a = V16_FROM_2xV8(al, ah);
        #pragma unroll
        for (int q = 0; q < 4; ++q) {
            v16h b = V16_FROM_2xV8(bl[q], bh[q]);
            acc[q] = __builtin_amdgcn_wmma_f32_16x16x32_f16(
                false, a, false, b, (short)0, acc[q], false, false);
        }
        al = nal; ah = nah;
        #pragma unroll
        for (int q = 0; q < 4; ++q) { bl[q] = nbl[q]; bh[q] = nbh[q]; }
    }
    // ---- epilogue: last K-step ----
    {
        v16h a = V16_FROM_2xV8(al, ah);
        #pragma unroll
        for (int q = 0; q < 4; ++q) {
            v16h b = V16_FROM_2xV8(bl[q], bh[q]);
            acc[q] = __builtin_amdgcn_wmma_f32_16x16x32_f16(
                false, a, false, b, (short)0, acc[q], false, false);
        }
    }

    // C/D layout: VGPR r -> row (r + (hi?8:0)), col = tile*16 + (lane&15)
    const int row0 = mt * 16 + (hi ? 8 : 0);
    #pragma unroll
    for (int q = 0; q < 4; ++q) {
        const int n = (nt4 * 4 + q) * 16 + r16;
        #pragma unroll
        for (int r = 0; r < 8; ++r)
            C[(size_t)(row0 + r) * Nc + n] = acc[q][r];
    }
}

// ---------------------------------------------------------------------------
// Per-node attention dots + softmax-state init.
//   as[n,h] = sum_c h[n,h,c]*a_src[h,c]   (64 channels per head)
// ---------------------------------------------------------------------------
template<int H>
__global__ void k_node_prep(const float* __restrict__ hfeat,
                            const float* __restrict__ a_src, const float* __restrict__ a_dst,
                            float* __restrict__ as, float* __restrict__ ad,
                            float* __restrict__ m, float* __restrict__ s, int N) {
    int idx = blockIdx.x * blockDim.x + threadIdx.x;
    if (idx >= N * H) return;
    int n = idx / H, hh = idx - n * H;
    const float* row = hfeat + (size_t)n * (H * 64) + hh * 64;
    const float* asr = a_src + hh * 64;
    const float* adr = a_dst + hh * 64;
    float vs = 0.f, vd = 0.f;
    #pragma unroll 8
    for (int c = 0; c < 64; ++c) { float x = row[c]; vs += x * asr[c]; vd += x * adr[c]; }
    as[idx] = vs;
    ad[idx] = vd;
    m[idx]  = -__builtin_inff();
    s[idx]  = 0.0f;
}

// ---------------------------------------------------------------------------
// Edge softmax passes. Edge id e < E: src=ei[e], dst=ei[E+e]; else self-loop.
// ---------------------------------------------------------------------------
__device__ __forceinline__ void edge_resolve(const int* ei, int E, int e, int& s, int& d) {
    if (e < E) { s = ei[e]; d = ei[E + e]; } else { s = d = e - E; }
}

template<int H>
__global__ void k_edge_max(const int* __restrict__ ei, int E, int N,
                           const float* __restrict__ as, const float* __restrict__ ad,
                           float* __restrict__ m) {
    int idx = blockIdx.x * blockDim.x + threadIdx.x;
    int total = (E + N) * H;
    if (idx >= total) return;
    int e = idx / H, h = idx - e * H;
    int s, d; edge_resolve(ei, E, e, s, d);
    float v = as[s * H + h] + ad[d * H + h];
    v = v > 0.f ? v : NEG_SLOPE * v;
    atomicMaxFloat(&m[d * H + h], v);
}

template<int H>
__global__ void k_edge_expsum(const int* __restrict__ ei, int E, int N,
                              const float* __restrict__ as, const float* __restrict__ ad,
                              const float* __restrict__ m,
                              float* __restrict__ wgt, float* __restrict__ ssum) {
    int idx = blockIdx.x * blockDim.x + threadIdx.x;
    int total = (E + N) * H;
    if (idx >= total) return;
    int e = idx / H, h = idx - e * H;
    int s, d; edge_resolve(ei, E, e, s, d);
    float v = as[s * H + h] + ad[d * H + h];
    v = v > 0.f ? v : NEG_SLOPE * v;
    float w = expf(v - m[d * H + h]);
    wgt[idx] = w;
    atomicAdd(&ssum[d * H + h], w);
}

template<int H>
__global__ void k_edge_norm(const int* __restrict__ ei, int E, int N,
                            const float* __restrict__ ssum, float* __restrict__ wgt) {
    int idx = blockIdx.x * blockDim.x + threadIdx.x;
    int total = (E + N) * H;
    if (idx >= total) return;
    int e = idx / H, h = idx - e * H;
    int s, d; edge_resolve(ei, E, e, s, d);
    wgt[idx] = wgt[idx] / ssum[d * H + h];
}

// Scatter-add messages: out[dst, t] += alpha[e, t>>6] * h[src, t], t in [0, H*64)
template<int H>
__global__ void k_edge_aggregate(const int* __restrict__ ei, int E, int N,
                                 const float* __restrict__ hfeat,
                                 const float* __restrict__ wgt,
                                 float* __restrict__ out) {
    const int TC = H * 64;
    unsigned idx = blockIdx.x * blockDim.x + threadIdx.x;
    unsigned total = (unsigned)(E + N) * TC;
    if (idx >= total) return;
    int e = (int)(idx / TC);
    int t = (int)(idx - (unsigned)e * TC);
    int s, d; edge_resolve(ei, E, e, s, d);
    int head = t >> 6;
    float coef = wgt[e * H + head];
    atomicAdd(&out[(size_t)d * TC + t], coef * hfeat[(size_t)s * TC + t]);
}

// ---------------------------------------------------------------------------
// ELU(agg + bias) -> f16 (input to layer-2 GEMM). cols = power of 2.
// ---------------------------------------------------------------------------
__global__ void k_elu_bias_f16(const float* __restrict__ agg, const float* __restrict__ b,
                               _Float16* __restrict__ out, int total, int colmask) {
    int i = blockIdx.x * blockDim.x + threadIdx.x;
    if (i >= total) return;
    float v = agg[i] + b[i & colmask];
    v = v > 0.f ? v : expm1f(v);
    out[i] = (_Float16)v;
}

__global__ void k_bias_add(float* __restrict__ out, const float* __restrict__ b,
                           int total, int colmask) {
    int i = blockIdx.x * blockDim.x + threadIdx.x;
    if (i >= total) return;
    out[i] += b[i & colmask];
}

// ---------------------------------------------------------------------------
// Host launch
// ---------------------------------------------------------------------------
static inline int ceil_div(long long a, int b) { return (int)((a + b - 1) / b); }

extern "C" void kernel_launch(void* const* d_in, const int* in_sizes, int n_in,
                              void* d_out, int out_size, void* d_ws, size_t ws_size,
                              hipStream_t stream) {
    const float* x      = (const float*)d_in[0];
    const int*   ei     = (const int*)  d_in[1];
    const float* W1     = (const float*)d_in[2];
    const float* a_src1 = (const float*)d_in[3];
    const float* a_dst1 = (const float*)d_in[4];
    const float* b1     = (const float*)d_in[5];
    const float* W2     = (const float*)d_in[6];
    const float* a_src2 = (const float*)d_in[7];
    const float* a_dst2 = (const float*)d_in[8];
    const float* b2     = (const float*)d_in[9];
    float* out = (float*)d_out;

    const int F_IN = 128, HID = 64, HEADS = 4, OUT = 64;
    const int F1 = HEADS * HID;                  // 256
    const int N = in_sizes[0] / F_IN;            // 50000 (divisible by 16)
    const int E = in_sizes[1] / 2;               // 800000
    const int EA = E + N;                        // edges + self loops

    // --- workspace carve-up ---
    char* base = (char*)d_ws;
    size_t off = 0;
    auto take = [&](size_t bytes) -> void* {
        void* p = base + off;
        off += (bytes + 255) & ~(size_t)255;
        return p;
    };
    _Float16* xh   = (_Float16*)take((size_t)N * F_IN * 2);
    _Float16* W1p  = (_Float16*)take((size_t)F_IN * F1 * 2);   // packed
    _Float16* W2p  = (_Float16*)take((size_t)F1 * OUT * 2);    // packed
    float* h1   = (float*)take((size_t)N * F1 * 4);
    float* as1  = (float*)take((size_t)N * HEADS * 4);
    float* ad1  = (float*)take((size_t)N * HEADS * 4);
    float* m1   = (float*)take((size_t)N * HEADS * 4);
    float* s1   = (float*)take((size_t)N * HEADS * 4);
    float* wgt1 = (float*)take((size_t)EA * HEADS * 4);
    float* agg1 = (float*)take((size_t)N * F1 * 4);
    _Float16* h2in = (_Float16*)take((size_t)N * F1 * 2);
    float* h2   = (float*)take((size_t)N * OUT * 4);
    float* as2  = (float*)take((size_t)N * 4);
    float* ad2  = (float*)take((size_t)N * 4);
    float* m2   = (float*)take((size_t)N * 4);
    float* s2   = (float*)take((size_t)N * 4);
    float* wgt2 = (float*)take((size_t)EA * 4);

    const int B = 256;

    // --- conversions / weight packing ---
    k_f32_to_f16<<<ceil_div((long long)N * F_IN, B), B, 0, stream>>>(x, xh, N * F_IN);
    k_pack_b_f16<<<ceil_div((long long)F_IN * F1, B), B, 0, stream>>>(W1, W1p, F_IN, F1);
    k_pack_b_f16<<<ceil_div((long long)F1 * OUT, B), B, 0, stream>>>(W2, W2p, F1, OUT);

    // ============ Layer 1: h1 = x @ W1  (WMMA, 16x64 tiles) ============
    {
        dim3 grid(N / 16, F1 / 64);
        k_wmma_gemm_f16<<<grid, 32, 0, stream>>>(xh, W1p, h1, N, F1, F_IN);
    }
    k_node_prep<4><<<ceil_div((long long)N * HEADS, B), B, 0, stream>>>(
        h1, a_src1, a_dst1, as1, ad1, m1, s1, N);
    k_zero_f32<<<ceil_div((long long)N * F1, B), B, 0, stream>>>(agg1, N * F1);

    k_edge_max<4><<<ceil_div((long long)EA * 4, B), B, 0, stream>>>(ei, E, N, as1, ad1, m1);
    k_edge_expsum<4><<<ceil_div((long long)EA * 4, B), B, 0, stream>>>(ei, E, N, as1, ad1, m1, wgt1, s1);
    k_edge_norm<4><<<ceil_div((long long)EA * 4, B), B, 0, stream>>>(ei, E, N, s1, wgt1);
    k_edge_aggregate<4><<<ceil_div((long long)EA * 256, B), B, 0, stream>>>(ei, E, N, h1, wgt1, agg1);

    // ELU(agg1 + b1) -> f16 input for layer 2
    k_elu_bias_f16<<<ceil_div((long long)N * F1, B), B, 0, stream>>>(agg1, b1, h2in, N * F1, F1 - 1);

    // ============ Layer 2: h2 = elu_out @ W2  (WMMA, 16x64 tiles) ============
    {
        dim3 grid(N / 16, OUT / 64);
        k_wmma_gemm_f16<<<grid, 32, 0, stream>>>(h2in, W2p, h2, N, OUT, F1);
    }
    k_node_prep<1><<<ceil_div((long long)N, B), B, 0, stream>>>(
        h2, a_src2, a_dst2, as2, ad2, m2, s2, N);
    k_zero_f32<<<ceil_div((long long)N * OUT, B), B, 0, stream>>>(out, N * OUT);

    k_edge_max<1><<<ceil_div((long long)EA, B), B, 0, stream>>>(ei, E, N, as2, ad2, m2);
    k_edge_expsum<1><<<ceil_div((long long)EA, B), B, 0, stream>>>(ei, E, N, as2, ad2, m2, wgt2, s2);
    k_edge_norm<1><<<ceil_div((long long)EA, B), B, 0, stream>>>(ei, E, N, s2, wgt2);
    k_edge_aggregate<1><<<ceil_div((long long)EA * 64, B), B, 0, stream>>>(ei, E, N, h2, wgt2, out);

    // heads=1 mean is identity; add bias b2
    k_bias_add<<<ceil_div((long long)N * OUT, B), B, 0, stream>>>(out, b2, N * OUT, OUT - 1);
}
